// DeepGCNGRUCell_71159018160981
// MI455X (gfx1250) — compile-verified, matrix operand order checked by reading, not run on hardware
//
#include <hip/hip_runtime.h>

// ---------------------------------------------------------------------------
// DeepGCN + 3x GRUCell for MI455X (gfx1250, wave32, WMMA).
// - All GEMM inputs (activations + weights) stored bf16 in memory (half the
//   streamed bytes on a BW-bound part); f32 accumulate via
//   v_wmma_f32_16x16x32_bf16; f32 epilogues/atomics/outputs.
// - LDS operand staging via GLOBAL_LOAD_ASYNC_TO_LDS_B128 (ASYNCcnt) with
//   padded rows for conflict-free ds_load_b128 fragment reads.
// ---------------------------------------------------------------------------

#define NN   100000   // nodes
#define NE   1600000  // edges
#define DH   128      // feature dim (D_IN == H == 128) == K for every GEMM
#define H3   384      // 3*H
#define KP   (DH + 8) // padded LDS row (bf16 elems): 272B stride, 16B aligned

#define USE_ASYNC_STAGING 1

typedef __attribute__((ext_vector_type(16))) __bf16 v16bf;
typedef __attribute__((ext_vector_type(8)))  float  v8f;

__device__ __forceinline__ unsigned short bfbits(float x) {
  union { __bf16 b; unsigned short s; } u;
  u.b = (__bf16)x;           // native RNE f32->bf16 (hardware cvt)
  return u.s;
}

#if USE_ASYNC_STAGING
// One wave-op copies 512B global -> LDS (16B per lane, per-lane LDS dest).
__device__ __forceinline__ void async_g2l_b128(unsigned ldsOff, const void* g) {
  asm volatile("global_load_async_to_lds_b128 %0, %1, off"
               :: "v"(ldsOff), "v"(g)
               : "memory");
}
__device__ __forceinline__ void wait_async0() {
  asm volatile("s_wait_asynccnt 0x0" ::: "memory");
}
#endif

// ---------------------------------------------------------------------------
// Utility kernels
// ---------------------------------------------------------------------------
__global__ void fill0_kernel(float* __restrict__ p, long n) {
  long i = (long)blockIdx.x * blockDim.x + threadIdx.x;
  if (i < n) p[i] = 0.0f;
}

__global__ void degree_kernel(const long long* __restrict__ dst,
                              float* __restrict__ deg, int e) {
  int i = blockIdx.x * blockDim.x + threadIdx.x;
  if (i < e) atomicAdd(&deg[(int)dst[i]], 1.0f);
}

__global__ void dinv_kernel(float* __restrict__ deg, int n) {
  int i = blockIdx.x * blockDim.x + threadIdx.x;
  if (i < n) deg[i] = rsqrtf(deg[i] + 1.0f);   // +1 self loop
}

// fp32 -> bf16 bulk convert, 4 elems/thread (b128 load, b64 store)
__global__ void f2bf_kernel(const float* __restrict__ in,
                            unsigned short* __restrict__ out, long n4) {
  long i = (long)blockIdx.x * blockDim.x + threadIdx.x;
  if (i >= n4) return;
  const float4 v = ((const float4*)in)[i];
  union { unsigned short s[4]; uint2 q; } pk;
  pk.s[0] = bfbits(v.x); pk.s[1] = bfbits(v.y);
  pk.s[2] = bfbits(v.z); pk.s[3] = bfbits(v.w);
  ((uint2*)out)[i] = pk.q;
}

// 128x128 transpose + convert: wt[n*128+k] = bf16(w[k*128+n])
__global__ void transpose128_bf_kernel(const float* __restrict__ w,
                                       unsigned short* __restrict__ wt) {
  int n = blockIdx.x;      // 128 blocks
  int k = threadIdx.x;     // 128 threads
  wt[n * DH + k] = bfbits(w[k * DH + n]);
}

// ---------------------------------------------------------------------------
// WMMA GEMM: C[M,Ncols] = A[M,128] @ Bt^T (+bias), A/Bt bf16 row-major,
// C fp32. Block = 256 thr (8 waves): one 16-row M-tile x one 128-col N-chunk.
// Stage A (16x256B) + B-chunk (128x256B) into LDS via async b128 copies
// (rows padded to 272B), then 4 unrolled k-steps of 2x ds_load_b128 per
// operand + v_wmma_f32_16x16x32_bf16 (bf16 frag layout maps vector elems
// 0..7 / 8..15 to contiguous K runs, so a 16B LDS read is half an operand).
// ---------------------------------------------------------------------------
__global__ __launch_bounds__(256)
void wmma_gemm_bt_kernel(const unsigned short* __restrict__ A,
                         const unsigned short* __restrict__ Bt,
                         const float* __restrict__ bias,
                         float* __restrict__ C,
                         int M, int Ncols, int relu) {
  __shared__ __align__(16) unsigned short lA[16 * KP];    // 4.25 KB
  __shared__ __align__(16) unsigned short lB[128 * KP];   // 34 KB

  const int tid    = threadIdx.x;
  const int lane   = tid & 31;
  const int wave   = tid >> 5;
  const int nchnk  = Ncols >> 7;              // N-chunks of 128
  const int mt     = blockIdx.x / nchnk;
  const int m0     = mt << 4;
  const int nbase  = (blockIdx.x % nchnk) << 7;

#if USE_ASYNC_STAGING
  {
    const unsigned lA0 = (unsigned)(size_t)&lA[0];   // LDS byte offset
    const unsigned lB0 = (unsigned)(size_t)&lB[0];
    const int l16  = lane & 15;
    const int rsel = lane >> 4;                      // lanes 0-15 / 16-31
    // A: 16 rows of 256B; each wave moves 2 rows in one op
    const int ar = (wave << 1) + rsel;
    async_g2l_b128(lA0 + (unsigned)(ar * (KP * 2) + l16 * 16),
                   A + (long)(m0 + ar) * DH + l16 * 8);
    // B: 128 rows; each wave moves 16 rows = 8 ops
#pragma unroll
    for (int i = 0; i < 8; ++i) {
      const int br = (wave << 4) + (i << 1) + rsel;
      async_g2l_b128(lB0 + (unsigned)(br * (KP * 2) + l16 * 16),
                     Bt + (long)(nbase + br) * DH + l16 * 8);
    }
    wait_async0();
  }
  __syncthreads();
#else
  // sync fallback: raw bf16 copy, 16B per thread-iteration
  for (int g = tid; g < 16 * 16; g += 256) {          // A: 16 rows x 16 chunks
    const int row = g >> 4, c = (g & 15) << 3;
    *(uint4*)&lA[row * KP + c] = *(const uint4*)(A + (long)(m0 + row) * DH + c);
  }
  for (int g = tid; g < 128 * 16; g += 256) {         // B: 128 rows
    const int row = g >> 4, c = (g & 15) << 3;
    *(uint4*)&lB[row * KP + c] =
        *(const uint4*)(Bt + (long)(nbase + row) * DH + c);
  }
  __syncthreads();
#endif

  // ---- compute: wave w owns N-tile [nbase + w*16, +16) ----
  const int half  = lane >> 4;
  const int l16   = lane & 15;
  const int brow0 = wave << 4;

  union Frag { uint4 q[2]; v16bf v; };
  v8f acc = {};
#pragma unroll
  for (int k0 = 0; k0 < DH; k0 += 32) {
    const int ko = k0 + half * 8;          // elems 0..7 -> K=ko..ko+7
    Frag fa, fb;
    fa.q[0] = *(const uint4*)&lA[l16 * KP + ko];          // ds_load_b128
    fa.q[1] = *(const uint4*)&lA[l16 * KP + ko + 16];     // elems 8..15
    fb.q[0] = *(const uint4*)&lB[(brow0 + l16) * KP + ko];
    fb.q[1] = *(const uint4*)&lB[(brow0 + l16) * KP + ko + 16];
    acc = __builtin_amdgcn_wmma_f32_16x16x32_bf16(
        /*neg_a=*/false, fa.v, /*neg_b=*/false, fb.v,
        /*c_mod=*/(short)0, acc, /*reuse_a=*/false, /*reuse_b=*/false);
  }

  const int n  = nbase + brow0 + l16;
  const float bb = bias ? bias[n] : 0.0f;
#pragma unroll
  for (int v = 0; v < 8; ++v) {
    const int m = m0 + v + half * 8;       // C/D VGPR layout
    float val = acc[v] + bb;
    if (relu) val = fmaxf(val, 0.0f);
    C[(long)m * Ncols + n] = val;
  }
}

// ---------------------------------------------------------------------------
// Edge scatter: one wave per edge, lane covers 4 channels (float4 gather,
// 4 fp32 atomics into agg[dst] -- agg (51 MB) lives in the 192 MB L2).
// ---------------------------------------------------------------------------
__global__ __launch_bounds__(256)
void edge_agg_kernel(const long long* __restrict__ src,
                     const long long* __restrict__ dst,
                     const float* __restrict__ dinv,
                     const float* __restrict__ xw,
                     float* __restrict__ agg, int e) {
  const int lane = threadIdx.x & 31;
  const int wave = threadIdx.x >> 5;
  const long idx = (long)blockIdx.x * 8 + wave;
  if (idx >= e) return;
  const int s = (int)src[idx];
  const int d = (int)dst[idx];
  const float norm = dinv[s] * dinv[d];
  const float4 v = *(const float4*)(xw + (long)s * DH + lane * 4);
  float* o = agg + (long)d * DH + lane * 4;
  atomicAdd(o + 0, v.x * norm);
  atomicAdd(o + 1, v.y * norm);
  atomicAdd(o + 2, v.z * norm);
  atomicAdd(o + 3, v.w * norm);
}

// outbf = bf16(relu(agg + xw*dinv^2 + b)); output feeds only the next GEMM.
__global__ void gcn_combine_kernel(const float* __restrict__ agg,
                                   const float* __restrict__ xw,
                                   const float* __restrict__ dinv,
                                   const float* __restrict__ b,
                                   unsigned short* __restrict__ outbf, long n) {
  long i = (long)blockIdx.x * blockDim.x + threadIdx.x;
  if (i >= n) return;
  const int row = (int)(i >> 7);
  const int col = (int)(i & 127);
  const float di = dinv[row];
  const float v = agg[i] + xw[i] * di * di + b[col];
  outbf[i] = bfbits(fmaxf(v, 0.0f));
}

// Fused GRU gate nonlinearity: gi/gh [N,3H] fp32, h [N,H] fp32.
// Writes fp32 result (final output) + bf16 copy (next layer's GEMM operand).
__global__ void gru_elem_kernel(const float* __restrict__ gi,
                                const float* __restrict__ gh,
                                const float* __restrict__ h,
                                float* __restrict__ out,
                                unsigned short* __restrict__ outbf, int nrows) {
  long i = (long)blockIdx.x * blockDim.x + threadIdx.x;
  const long total = (long)nrows * DH;
  if (i >= total) return;
  const long row = i >> 7;
  const int  c   = (int)(i & 127);
  const long b   = row * H3;
  const float ir = gi[b + c],         hr = gh[b + c];
  const float iz = gi[b + DH + c],    hz = gh[b + DH + c];
  const float in_ = gi[b + 2*DH + c], hn = gh[b + 2*DH + c];
  const float r = 1.0f / (1.0f + __expf(-(ir + hr)));
  const float z = 1.0f / (1.0f + __expf(-(iz + hz)));
  const float nn = tanhf(in_ + r * hn);
  const float res = (1.0f - z) * nn + z * h[i];
  out[i]   = res;
  outbf[i] = bfbits(res);
}

// ---------------------------------------------------------------------------
// Host-side orchestration (graph-capture safe: launches on `stream` only).
// ---------------------------------------------------------------------------
extern "C" void kernel_launch(void* const* d_in, const int* in_sizes, int n_in,
                              void* d_out, int out_size, void* d_ws, size_t ws_size,
                              hipStream_t stream) {
  (void)in_sizes; (void)n_in; (void)out_size; (void)ws_size;

  const float*     x    = (const float*)d_in[0];
  const long long* ei   = (const long long*)d_in[1];   // int64 [2,E]
  const float*     hin[3] = {(const float*)d_in[2], (const float*)d_in[3],
                             (const float*)d_in[4]};
  const float* gcn_w[2] = {(const float*)d_in[5], (const float*)d_in[7]};
  const float* gcn_b[2] = {(const float*)d_in[6], (const float*)d_in[8]};
  const float* gru_wih[3] = {(const float*)d_in[9],  (const float*)d_in[13], (const float*)d_in[17]};
  const float* gru_whh[3] = {(const float*)d_in[10], (const float*)d_in[14], (const float*)d_in[18]};
  const float* gru_bih[3] = {(const float*)d_in[11], (const float*)d_in[15], (const float*)d_in[19]};
  const float* gru_bhh[3] = {(const float*)d_in[12], (const float*)d_in[16], (const float*)d_in[20]};

  const long long* src = ei;
  const long long* dst = ei + NE;

  const long NH = (long)NN * DH;

  // ---- workspace carve (all chunks 16B aligned) -------------------------
  char* w = (char*)d_ws;
  float* deg = (float*)w;            w += (long)NN * 4;        // deg->dinv
  float* xw  = (float*)w;            w += NH * 4;              // GEMM out
  float* agg = (float*)w;            w += NH * 4;              // scatter acc
  float* gi  = (float*)w;            w += (long)NN * H3 * 4;
  float* gh  = (float*)w;            w += (long)NN * H3 * 4;
  unsigned short* xbf  = (unsigned short*)w; w += NH * 2;      // x / x1 / x2 bf16
  unsigned short* hbf[3];
  for (int i = 0; i < 3; ++i) { hbf[i] = (unsigned short*)w; w += NH * 2; }
  unsigned short* wtb  = (unsigned short*)w; w += DH * DH * 2; // gcn W^T bf16
  unsigned short* wihb = (unsigned short*)w; w += H3 * DH * 2;
  unsigned short* whhb = (unsigned short*)w; w += H3 * DH * 2;

  float* h1n = (float*)d_out;
  float* h2n = h1n + NH;
  float* h3n = h2n + NH;

  const dim3 blk(256);
  const int gemm_blocks_128 = (NN / 16) * (DH / 128);     // 6250
  const int gemm_blocks_384 = (NN / 16) * (H3 / 128);     // 18750
  const int edge_blocks     = NE / 8;                     // 200000
  const int elem_blocks     = (int)((NH + 255) / 256);    // 50000
  const int cvt_blocks      = (int)((NH / 4 + 255) / 256);
  const int cvtw_blocks     = (H3 * DH / 4 + 255) / 256;

  // --- degrees -> dinv ---
  fill0_kernel<<<(NN + 255) / 256, blk, 0, stream>>>(deg, NN);
  degree_kernel<<<(NE + 255) / 256, blk, 0, stream>>>(dst, deg, NE);
  dinv_kernel<<<(NN + 255) / 256, blk, 0, stream>>>(deg, NN);

  // --- bf16 copies of GEMM A-side inputs ---
  f2bf_kernel<<<cvt_blocks, blk, 0, stream>>>(x, xbf, NH / 4);
  for (int i = 0; i < 3; ++i)
    f2bf_kernel<<<cvt_blocks, blk, 0, stream>>>(hin[i], hbf[i], NH / 4);

  // --- GCN layers: xbf -> xw -> agg -> xbf (in place, elementwise) ---
  for (int l = 0; l < 2; ++l) {
    transpose128_bf_kernel<<<DH, DH, 0, stream>>>(gcn_w[l], wtb);
    wmma_gemm_bt_kernel<<<gemm_blocks_128, blk, 0, stream>>>(
        xbf, wtb, nullptr, xw, NN, DH, 0);
    fill0_kernel<<<elem_blocks, blk, 0, stream>>>(agg, NH);
    edge_agg_kernel<<<edge_blocks, blk, 0, stream>>>(src, dst, deg, xw, agg, NE);
    gcn_combine_kernel<<<elem_blocks, blk, 0, stream>>>(agg, xw, deg, gcn_b[l],
                                                        xbf, NH);
  }

  // --- GRU chain: (x2,h1)->h1n, (h1n,h2)->h2n, (h2n,h3)->h3n ---
  // A-side bf16 operands; gru_elem overwrites hbf[i] with bf16(h_i_new),
  // which is exactly the next layer's A operand.
  const unsigned short* gru_xbf[3] = {xbf, hbf[0], hbf[1]};
  float* gru_o[3] = {h1n, h2n, h3n};
  for (int i = 0; i < 3; ++i) {
    f2bf_kernel<<<cvtw_blocks, blk, 0, stream>>>(gru_wih[i], wihb,
                                                 (long)H3 * DH / 4);
    f2bf_kernel<<<cvtw_blocks, blk, 0, stream>>>(gru_whh[i], whhb,
                                                 (long)H3 * DH / 4);
    wmma_gemm_bt_kernel<<<gemm_blocks_384, blk, 0, stream>>>(
        gru_xbf[i], wihb, gru_bih[i], gi, NN, H3, 0);
    wmma_gemm_bt_kernel<<<gemm_blocks_384, blk, 0, stream>>>(
        hbf[i], whhb, gru_bhh[i], gh, NN, H3, 0);
    gru_elem_kernel<<<elem_blocks, blk, 0, stream>>>(gi, gh, hin[i],
                                                     gru_o[i], hbf[i], NN);
  }
}
